// GeneralMACE_40922448396307
// MI455X (gfx1250) — matrix operand order repaired and checked by reading, not compiled, not run on hardware
//
#include <hip/hip_runtime.h>

// ---------------- types ----------------
typedef __bf16 bf16_t;
typedef __attribute__((ext_vector_type(16))) bf16_t v16bf;
typedef __attribute__((ext_vector_type(2)))  bf16_t bf16x2;
typedef __attribute__((ext_vector_type(8)))  float  v8f;

#define NN 32768
#define EE 524288

__device__ __forceinline__ bf16_t f2bf(float x) {
  union { float f; unsigned u; } a; a.f = x;
  unsigned r = a.u + 0x7FFFu + ((a.u >> 16) & 1u);   // round-to-nearest-even
  unsigned short h = (unsigned short)(r >> 16);
  bf16_t o; __builtin_memcpy(&o, &h, 2); return o;
}

// Native CDNA5 f32 scatter-add: GLOBAL_ATOMIC_ADD_F32, device scope so the
// add is coherent across WGPs (L2 atomic units), no return (STOREcnt only;
// S_ENDPGM performs an implicit wait-idle).
__device__ __forceinline__ void atomAddF(float* p, float v) {
  asm volatile("global_atomic_add_f32 %0, %1, off scope:SCOPE_DEV"
               :: "v"(p), "v"(v) : "memory");
}

__device__ __forceinline__ v8f wmma_bf16(v16bf a, v16bf b, v8f c) {
  // (neg_a, A, neg_b, B, c_mod, C, reuse_a, reuse_b)
  return __builtin_amdgcn_wmma_f32_16x16x32_bf16(false, a, false, b, (short)0, c, false, false);
}

// ---------------- weight pack: W[K][N] f32 -> WT[N][Kpad] bf16 (zero pad) ----------------
__global__ __launch_bounds__(256) void k_packWT(const float* __restrict__ W,
                                                bf16_t* __restrict__ WT,
                                                int K, int Ncols, int Kpad) {
  int i = blockIdx.x * 256 + threadIdx.x;
  if (i >= Ncols * Kpad) return;
  int g = i / Kpad, f = i % Kpad;
  WT[i] = f2bf(f < K ? W[f * Ncols + g] : 0.0f);
}

// ---------------- per-edge geometry: Y1 (l=1 SH) and bessel*envelope (bf16) ----------------
__global__ __launch_bounds__(256) void k_edge_geom(const float* __restrict__ vec,
                                                   float* __restrict__ Y1,
                                                   bf16_t* __restrict__ Rbf) {
  int e = blockIdx.x * 256 + threadIdx.x;
  if (e >= EE) return;
  float vx = vec[3*e+0], vy = vec[3*e+1], vz = vec[3*e+2];
  float r = sqrtf(vx*vx + vy*vy + vz*vz + 1e-12f);
  float inv = 1.0f / r;
  const float s3 = 1.7320508075688772f;
  Y1[3*e+0] = s3*vx*inv; Y1[3*e+1] = s3*vy*inv; Y1[3*e+2] = s3*vz*inv;
  float u = r * 0.2f;                      // r / RMAX
  float env = 0.f;
  if (u < 1.f) {
    float u2 = u*u, u3 = u2*u, u6 = u3*u3, u7 = u6*u, u8 = u7*u;
    env = 1.f - 28.f*u6 + 48.f*u7 - 21.f*u8;
  }
  float pref = 0.6324555320336759f * inv * env;   // sqrt(2/RMAX)/r * env
#pragma unroll
  for (int n = 1; n <= 8; ++n)
    Rbf[8*e + n - 1] = f2bf(pref * __sinf((float)n * 0.6283185307179586f * r));
}

// ---------------- embedding ----------------
__global__ __launch_bounds__(64) void k_embed(const float* __restrict__ Wemb,
                                              const int* __restrict__ spec,
                                              float* __restrict__ x0) {
  int n = blockIdx.x, f = threadIdx.x;
  x0[(size_t)n*64 + f] = Wemb[spec[n]*64 + f];
}

// ---------------- C[M,64] = scale * (A[M,64] @ W) with W packed as WT[64g][64f] bf16 ----------------
__global__ __launch_bounds__(128) void k_gemm64(const float* __restrict__ A,
                                                const bf16_t* __restrict__ WT,
                                                float* __restrict__ C, float scale) {
  __shared__ __align__(16) bf16_t As[16][64];
  int tid = threadIdx.x, lane = tid & 31, wave = tid >> 5;
  int row0 = blockIdx.x * 16;
  for (int i = tid; i < 16*64; i += 128) {
    int r = i >> 6, c = i & 63;
    As[r][c] = f2bf(A[(size_t)(row0 + r)*64 + c]);
  }
  __syncthreads();
  int m = lane & 15, half = lane >> 4;
  v8f acc = {};
#pragma unroll
  for (int ks = 0; ks < 2; ++ks) {
    int kb = ks * 32;
    v16bf a, b;
#pragma unroll
    for (int v = 0; v < 8; ++v) {
      int ka = kb + ((v >> 2) << 4) + (half << 3) + ((v & 3) << 1);
      bf16x2 pa = *(const bf16x2*)&As[m][ka];
      a[2*v] = pa.x; a[2*v+1] = pa.y;
      int kbb = kb + ((v >> 2) << 4) + (half << 3) + ((v & 3) << 1);
      bf16x2 pb = *(const bf16x2*)&WT[(size_t)(wave*16 + m)*64 + kbb];
      b[2*v] = pb.x; b[2*v+1] = pb.y;
    }
    acc = wmma_bf16(a, b, acc);
  }
  int col = wave*16 + m;
#pragma unroll
  for (int r = 0; r < 8; ++r)
    C[(size_t)(row0 + half*8 + r)*64 + col] = acc[r] * scale;
}

// ---------------- per-node species-indexed 64x64 matmul: C[n] = X[n] @ Wsp[spec[n]] ----------------
__global__ __launch_bounds__(64) void k_species_mm(const float* __restrict__ X,
                                                   const float* __restrict__ Wsp,
                                                   const int* __restrict__ spec,
                                                   float* __restrict__ C) {
  int n = blockIdx.x, g = threadIdx.x;
  const float* W  = Wsp + (size_t)spec[n] * 4096;
  const float* xr = X + (size_t)n * 64;
  float acc = 0.f;
#pragma unroll 8
  for (int f = 0; f < 64; ++f) acc = fmaf(xr[f], W[f*64 + g], acc);
  C[(size_t)n*64 + g] = acc;
}

// ---------------- fused edge kernel: radial MLP (WMMA) + CG tensor product + scatter ----------------
__global__ __launch_bounds__(128) void k_edge_tp(
    const bf16_t* __restrict__ Rbf,      // [E][8]
    const float*  __restrict__ Y1,       // [E][3]
    const bf16_t* __restrict__ Wr1T,     // [64][32] (K=8 zero-padded)
    const bf16_t* __restrict__ Wr2T,     // [320][64]
    const float*  __restrict__ h0,       // [N][64]
    const float*  __restrict__ h1,       // [3][N][64]
    const int*    __restrict__ senders,
    const int*    __restrict__ receivers,
    float* __restrict__ a0,              // [N][64]  (atomic accumulate)
    float* __restrict__ a1)              // [3][N][64]
{
  __shared__ __align__(16) bf16_t hidA[16][64];
  __shared__ float wbuf[16][5][64];
  __shared__ float y1s[16][3];
  __shared__ int   snd[16], rcv[16];

  int tid = threadIdx.x, lane = tid & 31, wave = tid >> 5;
  int e0 = blockIdx.x * 16;
  if (tid < 16) { snd[tid] = senders[e0+tid]; rcv[tid] = receivers[e0+tid]; }
  if (tid < 48) { y1s[tid/3][tid%3] = Y1[(size_t)e0*3 + tid]; }
  int m = lane & 15, half = lane >> 4;

  // GEMM1: hidden = silu(R @ Wr1), K padded 8->32 (upper K and upper lane-half are zero)
  {
    v16bf a = {};
    if (half == 0) {
      const bf16_t* rrow = Rbf + (size_t)(e0 + m) * 8;
#pragma unroll
      for (int v = 0; v < 4; ++v) {     // K = 0..7 only
        bf16x2 p = *(const bf16x2*)(rrow + 2*v);
        a[2*v] = p.x; a[2*v+1] = p.y;
      }
    }
    v16bf b;
    int gcol = wave*16 + m;
#pragma unroll
    for (int v = 0; v < 8; ++v) {
      int k = (half << 4) + (v << 1);
      bf16x2 p = *(const bf16x2*)&Wr1T[(size_t)gcol*32 + k];
      b[2*v] = p.x; b[2*v+1] = p.y;
    }
    v8f acc = {};
    acc = wmma_bf16(a, b, acc);
#pragma unroll
    for (int r = 0; r < 8; ++r) {
      float x = acc[r];
      hidA[half*8 + r][gcol] = f2bf(x / (1.f + __expf(-x)));   // silu
    }
  }
  __syncthreads();

  // GEMM2: w[16e,320] = hidden @ Wr2 ; each wave does 5 of the 20 16-col tiles
  for (int t = 0; t < 5; ++t) {
    int nt = wave*5 + t;
    int gcol = nt*16 + m;
    v8f acc = {};
#pragma unroll
    for (int ks = 0; ks < 2; ++ks) {
      int kb = ks * 32;
      v16bf a, b;
#pragma unroll
      for (int v = 0; v < 8; ++v) {
        int ka = kb + ((v >> 2) << 4) + (half << 3) + ((v & 3) << 1);
        bf16x2 pa = *(const bf16x2*)&hidA[m][ka];
        a[2*v] = pa.x; a[2*v+1] = pa.y;
        int kbb = kb + ((v >> 2) << 4) + (half << 3) + ((v & 3) << 1);
        bf16x2 pb = *(const bf16x2*)&Wr2T[(size_t)gcol*64 + kbb];
        b[2*v] = pb.x; b[2*v+1] = pb.y;
      }
      acc = wmma_bf16(a, b, acc);
    }
    int p = gcol >> 6, f = gcol & 63;
#pragma unroll
    for (int r = 0; r < 8; ++r)
      wbuf[half*8 + r][p][f] = acc[r];
  }
  __syncthreads();

  // messages + scatter: 128 threads = 64 channels x 2 edge slots, 8 rounds
  int f = tid & 63, esub = tid >> 6;
  const size_t pst = (size_t)NN * 64;
  for (int er = 0; er < 8; ++er) {
    int el = er*2 + esub;
    int s  = snd[el], rc = rcv[el];
    float yx = y1s[el][0], yy = y1s[el][1], yz = y1s[el][2];
    size_t sb = (size_t)s*64 + f;
    float s0  = h0[sb];
    float sx  = h1[sb], sy = h1[sb + pst], sz = h1[sb + 2*pst];
    float w0 = wbuf[el][0][f], w1 = wbuf[el][1][f], w2 = wbuf[el][2][f],
          w3 = wbuf[el][3][f], w4 = wbuf[el][4][f];
    float dot = sx*yx + sy*yy + sz*yz;
    float m0v = w0*s0 + w1*dot;
    float cx = sy*yz - sz*yy, cy = sz*yx - sx*yz, cz = sx*yy - sy*yx;  // s1 x Y1
    float mx = w2*sx + w3*s0*yx + w4*cx;
    float my = w2*sy + w3*s0*yy + w4*cy;
    float mz = w2*sz + w3*s0*yz + w4*cz;
    size_t rb = (size_t)rc*64 + f;
    atomAddF(&a0[rb], m0v);
    atomAddF(&a1[rb], mx);
    atomAddF(&a1[rb + pst], my);
    atomAddF(&a1[rb + 2*pst], mz);
  }
}

// ---------------- product basis + skip + readout ----------------
__global__ __launch_bounds__(64) void k_product(
    const float* __restrict__ A0, const float* __restrict__ A1,
    const float* __restrict__ SC0, const float* __restrict__ SC1,
    const float* __restrict__ Wp0a, const float* __restrict__ Wp0b, const float* __restrict__ Wp0c,
    const float* __restrict__ Wp1a, const float* __restrict__ Wp1b,
    const int* __restrict__ spec,
    float* __restrict__ x0, float* __restrict__ x1,
    const float* __restrict__ Wro,    // layer0 readout [64], or null
    const float* __restrict__ Wm1,    // final MLP [64][16], or null
    const float* __restrict__ Wm2,    // final MLP [16], or null
    float* __restrict__ out, int outCol)
{
  __shared__ float xs[64];
  __shared__ float red[64];
  int n = blockIdx.x, f = threadIdx.x;
  int sp = spec[n];
  size_t base = (size_t)n*64 + f;
  const size_t pst = (size_t)NN * 64;
  float a0v = A0[base];
  float ax = A1[base], ay = A1[base + pst], az = A1[base + 2*pst];
  float w0a = Wp0a[sp*64+f], w0b = Wp0b[sp*64+f], w0c = Wp0c[sp*64+f];
  float w1a = Wp1a[sp*64+f], w1b = Wp1b[sp*64+f];
  float X0 = w0a*a0v + w0b*a0v*a0v + w0c*(ax*ax + ay*ay + az*az);
  float c1 = w1a + w1b*a0v;
  float Xx = c1*ax, Xy = c1*ay, Xz = c1*az;
  if (SC0) {
    X0 += SC0[base];
    Xx += SC1[base]; Xy += SC1[base + pst]; Xz += SC1[base + 2*pst];
  }
  x0[base] = X0;
  x1[base] = Xx; x1[base + pst] = Xy; x1[base + 2*pst] = Xz;
  if (Wro) {
    red[f] = X0 * Wro[f];
    __syncthreads();
    for (int s = 32; s > 0; s >>= 1) { if (f < s) red[f] += red[f+s]; __syncthreads(); }
    if (f == 0) out[(size_t)n*2 + outCol] = red[0];
  } else {
    xs[f] = X0;
    __syncthreads();
    if (f < 16) {
      float h = 0.f;
#pragma unroll 8
      for (int k = 0; k < 64; ++k) h = fmaf(xs[k], Wm1[k*16 + f], h);
      red[f] = (h / (1.f + __expf(-h))) * Wm2[f];
    }
    __syncthreads();
    if (f == 0) {
      float s = 0.f;
#pragma unroll
      for (int k = 0; k < 16; ++k) s += red[k];
      out[(size_t)n*2 + outCol] = s;
    }
  }
}

// ---------------- host ----------------
extern "C" void kernel_launch(void* const* d_in, const int* in_sizes, int n_in,
                              void* d_out, int out_size, void* d_ws, size_t ws_size,
                              hipStream_t stream) {
  (void)n_in; (void)out_size; (void)ws_size;
  const float* vectors   = (const float*)d_in[0];
  const int*   spec      = (const int*)  d_in[1];
  const int*   senders   = (const int*)  d_in[2];
  const int*   receivers = (const int*)  d_in[3];
  const float* Wemb      = (const float*)d_in[4];

  // Param index maps. Slot order per layer:
  // 0 Wup0, 1 Wup1, 2 Wr1, 3 Wr2, 4 Wd0, 5 Wd1, 6 Wp0a, 7 Wp0b, 8 Wp0c,
  // 9 Wp1a, 10 Wp1b, 11 Wfirst0/Wsk0, 12 Wfirst1/Wsk1, 13 Wro/Wm1, 14 Wm2
  // jax tree_flatten sorts dict keys; a plain values() flatten keeps insertion order.
  // Disambiguate: in_sizes[7]==40960 -> sorted (Wfirst0), ==512 -> insertion (Wr1).
  int idx[2][15];
  if (in_sizes[7] == 40960) {   // sorted
    int l0[15] = {17,18,14,15, 5, 6, 9,10,11,12,13, 7, 8,16,-1};
    int l1[15] = {32,33,28,29,19,20,23,24,25,26,27,30,31,21,22};
    for (int k = 0; k < 15; ++k) { idx[0][k] = l0[k]; idx[1][k] = l1[k]; }
  } else {                      // insertion
    int l0[15] = { 5, 6, 7, 8, 9,10,11,12,13,14,15,16,17,18,-1};
    int l1[15] = {19,20,21,22,23,24,25,26,27,28,29,30,31,32,33};
    for (int k = 0; k < 15; ++k) { idx[0][k] = l0[k]; idx[1][k] = l1[k]; }
  }
  auto P = [&](int l, int k) -> const float* {
    int i = idx[l][k];
    return (i < 0) ? nullptr : (const float*)d_in[i];
  };

  // workspace bump allocator
  char* ws = (char*)d_ws; size_t off = 0;
  auto alloc = [&](size_t bytes) -> char* {
    char* p = ws + off; off += (bytes + 255) & ~(size_t)255; return p;
  };
  const size_t pst = (size_t)NN * 64;
  float*  Y1  = (float*) alloc((size_t)EE*3*4);
  bf16_t* Rbf = (bf16_t*)alloc((size_t)EE*8*2);
  float*  x0  = (float*) alloc(pst*4);
  float*  x1  = (float*) alloc(pst*4*3);
  float*  h0  = (float*) alloc(pst*4);
  float*  h1  = (float*) alloc(pst*4*3);
  float*  a0  = (float*) alloc(pst*4);
  float*  a1  = (float*) alloc(pst*4*3);
  float*  sc0 = (float*) alloc(pst*4);
  float*  sc1 = (float*) alloc(pst*4*3);
  bf16_t *Wup0T[2], *Wup1T[2], *Wd0T[2], *Wd1T[2], *Wr1Tp[2], *Wr2Tp[2];
  for (int l = 0; l < 2; ++l) {
    Wup0T[l] = (bf16_t*)alloc(64*64*2);
    Wup1T[l] = (bf16_t*)alloc(64*64*2);
    Wd0T[l]  = (bf16_t*)alloc(64*64*2);
    Wd1T[l]  = (bf16_t*)alloc(64*64*2);
    Wr1Tp[l] = (bf16_t*)alloc(64*32*2);
    Wr2Tp[l] = (bf16_t*)alloc(320*64*2);
  }

  // pack weights to bf16 transposed layouts
  for (int l = 0; l < 2; ++l) {
    k_packWT<<<(64*64+255)/256, 256, 0, stream>>>(P(l,0), Wup0T[l], 64, 64, 64);
    k_packWT<<<(64*64+255)/256, 256, 0, stream>>>(P(l,1), Wup1T[l], 64, 64, 64);
    k_packWT<<<(64*32+255)/256, 256, 0, stream>>>(P(l,2), Wr1Tp[l],  8, 64, 32);
    k_packWT<<<(320*64+255)/256,256, 0, stream>>>(P(l,3), Wr2Tp[l], 64,320, 64);
    k_packWT<<<(64*64+255)/256, 256, 0, stream>>>(P(l,4), Wd0T[l],  64, 64, 64);
    k_packWT<<<(64*64+255)/256, 256, 0, stream>>>(P(l,5), Wd1T[l],  64, 64, 64);
  }

  k_edge_geom<<<EE/256, 256, 0, stream>>>(vectors, Y1, Rbf);
  k_embed<<<NN, 64, 0, stream>>>(Wemb, spec, x0);
  hipMemsetAsync(x1, 0, pst*4*3, stream);

  float* out = (float*)d_out;
  const float down_scale = 0.25f;   // 1/sqrt(AVG=16)

  for (int l = 0; l < 2; ++l) {
    if (l > 0) {   // species-indexed skip from pre-update features
      k_species_mm<<<NN, 64, 0, stream>>>(x0, P(l,11), spec, sc0);
      for (int d = 0; d < 3; ++d)
        k_species_mm<<<NN, 64, 0, stream>>>(x1 + d*pst, P(l,12), spec, sc1 + d*pst);
    }
    hipMemsetAsync(a0, 0, pst*4,   stream);
    hipMemsetAsync(a1, 0, pst*4*3, stream);
    // linear up
    k_gemm64<<<NN/16, 128, 0, stream>>>(x0, Wup0T[l], h0, 1.0f);
    for (int d = 0; d < 3; ++d)
      k_gemm64<<<NN/16, 128, 0, stream>>>(x1 + d*pst, Wup1T[l], h1 + d*pst, 1.0f);
    // fused radial MLP + tensor product + scatter
    k_edge_tp<<<EE/16, 128, 0, stream>>>(Rbf, Y1, Wr1Tp[l], Wr2Tp[l],
                                         h0, h1, senders, receivers, a0, a1);
    // linear down (reuse h as output; h already consumed)
    k_gemm64<<<NN/16, 128, 0, stream>>>(a0, Wd0T[l], h0, down_scale);
    for (int d = 0; d < 3; ++d)
      k_gemm64<<<NN/16, 128, 0, stream>>>(a1 + d*pst, Wd1T[l], h1 + d*pst, down_scale);
    const float *A0p = h0, *A1p = h1;
    if (l == 0) {  // skip_tp_first: species-indexed linear (reuse a as output)
      k_species_mm<<<NN, 64, 0, stream>>>(h0, P(l,11), spec, a0);
      for (int d = 0; d < 3; ++d)
        k_species_mm<<<NN, 64, 0, stream>>>(h1 + d*pst, P(l,12), spec, a1 + d*pst);
      A0p = a0; A1p = a1;
    }
    k_product<<<NN, 64, 0, stream>>>(A0p, A1p,
        (l > 0) ? sc0 : nullptr, (l > 0) ? sc1 : nullptr,
        P(l,6), P(l,7), P(l,8), P(l,9), P(l,10), spec,
        x0, x1,
        (l == 0) ? P(l,13) : nullptr,            // Wro
        (l == 1) ? P(l,13) : nullptr,            // Wm1
        (l == 1) ? P(l,14) : nullptr,            // Wm2
        out, l);
  }
}
